// ST_37331855737369
// MI455X (gfx1250) — matrix operand order repaired and checked by reading, not compiled, not run on hardware
//
#include <hip/hip_runtime.h>
#include <math.h>

// ---------------- problem constants ----------------
#define BB 8
#define TT 20
#define NN 64
#define HD 128
#define BTN (BB*TT*NN)        // 10240
#define BTNN (BTN*NN)         // 655360

typedef _Float16 v16h __attribute__((ext_vector_type(16)));
typedef _Float16 h8   __attribute__((ext_vector_type(8)));
typedef float    v8f  __attribute__((ext_vector_type(8)));

// ---------------- WMMA fragment loads ----------------
// A (16x32 f16): lanes 0-15 = rows M=0..15 holding K 0..7 (elems0-7) and K 16..23 (elems8-15)
//                lanes16-31 = rows M=0..15 holding K 8..15 and K 24..31
__device__ __forceinline__ v16h load_a_frag(const _Float16* p, int hs) {
    union { v16h v; h8 h[2]; } u;
    u.h[0] = *(const h8*)(p + hs * 8);
    u.h[1] = *(const h8*)(p + 16 + hs * 8);
    return u.v;
}
// B (32x16 f16): lanes 0-15 = cols N=0..15 holding K 0..7 and K 16..23; lanes16-31 = K 8..15 and K 24..31
__device__ __forceinline__ v16h load_b_frag(const _Float16* p, int hs) {
    union { v16h v; h8 h[2]; } u;
    u.h[0] = *(const h8*)(p + hs * 8);
    u.h[1] = *(const h8*)(p + 16 + hs * 8);
    return u.v;
}

// ---------------- weight convert/transpose: W[K][Nc] f32 -> Bt[Nc][K] f16 ----------------
__global__ void k_cvt(const float* __restrict__ src, _Float16* __restrict__ dst,
                      int K, int Nc) {
    int idx = blockIdx.x * 256 + threadIdx.x;
    if (idx >= K * Nc) return;
    int k = idx / Nc, n = idx % Nc;
    dst[n * K + k] = (_Float16)src[idx];
}

// ---------------- first MLP layer (3 -> 128) + LN + ReLU for x ----------------
__global__ void k_fc1_x(const float* __restrict__ af, const float* __restrict__ w1,
                        const float* __restrict__ b1, const float* __restrict__ g,
                        const float* __restrict__ be, _Float16* __restrict__ out) {
    __shared__ float s1[128], s2[128];
    int row = blockIdx.x;
    int c = threadIdx.x;
    const float* a = af + (size_t)row * 3;
    float acc = b1[c] + a[0] * w1[c] + a[1] * w1[128 + c] + a[2] * w1[256 + c];
    s1[c] = acc; s2[c] = acc * acc;
    __syncthreads();
    for (int off = 64; off > 0; off >>= 1) {
        if (c < off) { s1[c] += s1[c + off]; s2[c] += s2[c + off]; }
        __syncthreads();
    }
    float m  = s1[0] * (1.f / 128.f);
    float vr = s2[0] * (1.f / 128.f) - m * m;
    float rs = rsqrtf(vr + 1e-5f);
    float v = (acc - m) * rs * g[c] + be[c];
    out[(size_t)row * 128 + c] = (_Float16)fmaxf(v, 0.f);
}

// ---------------- first MLP layer (5 -> 128) + LN + ReLU for kv (features built on the fly) --
__global__ void k_fc1_kv(const float* __restrict__ rf, const float* __restrict__ af,
                         const float* __restrict__ w1, const float* __restrict__ b1,
                         const float* __restrict__ g, const float* __restrict__ be,
                         _Float16* __restrict__ out) {
    __shared__ float s1[128], s2[128];
    int r = blockIdx.x;                 // row in [0, BTNN)
    int c = threadIdx.x;
    int j = r & 63;
    int i = (r >> 6) & 63;
    int t = (r >> 12) % TT;
    int b = r / (TT * NN * NN);
    // rot_feats [B,N,T,2]; rot(b,t,n) = rf[((b*N+n)*T+t)*2 + {0,1}]
    float e0 = rf[((b * NN + j) * TT + t) * 2 + 0] - rf[((b * NN + i) * TT + t) * 2 + 0];
    float e1 = rf[((b * NN + j) * TT + t) * 2 + 1] - rf[((b * NN + i) * TT + t) * 2 + 1];
    const float* ag = af + (size_t)(((b * TT + t) * NN) + j) * 3;
    float acc = b1[c] + e0 * w1[c] + e1 * w1[128 + c] + ag[0] * w1[256 + c]
                      + ag[1] * w1[384 + c] + ag[2] * w1[512 + c];
    s1[c] = acc; s2[c] = acc * acc;
    __syncthreads();
    for (int off = 64; off > 0; off >>= 1) {
        if (c < off) { s1[c] += s1[c + off]; s2[c] += s2[c + off]; }
        __syncthreads();
    }
    float m  = s1[0] * (1.f / 128.f);
    float vr = s2[0] * (1.f / 128.f) - m * m;
    float rs = rsqrtf(vr + 1e-5f);
    float v = (acc - m) * rs * g[c] + be[c];
    out[(size_t)r * 128 + c] = (_Float16)fmaxf(v, 0.f);
}

// ---------------- generic WMMA GEMM with fused epilogue ----------------
// C[M,Nc] = A[M,K](f16) x Bt[Nc,K](f16)^T + bias; optional +resid, LayerNorm(g,be), ReLU.
// grid = (M/16, Nc/128), block = 256 (8 waves, each wave one 16x16 tile).
// The shared 16-row A block is staged into LDS once via CDNA5 async global->LDS
// copies (ASYNCcnt), then all 8 waves read fragments via ds_load_b128.
// LN only legal when Nc==128 (gridDim.y==1) -- caller guarantees.
__global__ void k_gemm(const _Float16* __restrict__ A, const _Float16* __restrict__ Bt,
                       const float* __restrict__ bias, const float* __restrict__ resid,
                       const float* __restrict__ g, const float* __restrict__ be,
                       int relu, int nt,
                       float* __restrict__ out32, _Float16* __restrict__ out16,
                       int Nc, int K) {
    __shared__ _Float16 ashare[16 * 264];   // 16 rows x (K+8) halves, K<=256
    __shared__ float tile[16 * 132];
    __shared__ float sm[16], sr[16];
    int wave = threadIdx.x >> 5;
    int lane = threadIdx.x & 31;
    int hs   = lane >> 4;
    int mr   = lane & 15;
    size_t row0 = (size_t)blockIdx.x * 16;
    int    col0 = blockIdx.y * 128;
    const int KS = K + 8;                   // padded LDS row stride (halves)

    // ---- async global -> LDS staging of the A block (16 x K halves) ----
    {
        unsigned abase = (unsigned)(size_t)(&ashare[0]);   // low 32 bits = LDS offset
        int kc = K >> 3;                                   // 16B chunks per row
        int chunks = 16 * kc;
        for (int cid = threadIdx.x; cid < chunks; cid += 256) {
            int row = cid / kc;
            int c8  = cid - row * kc;
            const _Float16* gp = A + (row0 + row) * (size_t)K + c8 * 8;
            unsigned lds = abase + (unsigned)(row * KS + c8 * 8) * 2u;
            unsigned long long ga = (unsigned long long)(size_t)gp;
            asm volatile("global_load_async_to_lds_b128 %0, %1, off"
                         :: "v"(lds), "v"(ga) : "memory");
        }
#if __has_builtin(__builtin_amdgcn_s_wait_asynccnt)
        __builtin_amdgcn_s_wait_asynccnt(0);
#else
        asm volatile("s_wait_asynccnt 0" ::: "memory");
#endif
    }
    __syncthreads();

    const _Float16* Al = &ashare[mr * KS];
    const _Float16* Bp = Bt + (size_t)(col0 + wave * 16 + mr) * K;

    v8f acc = {};
    for (int k0 = 0; k0 < K; k0 += 32) {
        v16h a = load_a_frag(Al + k0, hs);   // ds_load_b128 x2
        v16h b = load_b_frag(Bp + k0, hs);   // global_load_b128 x2 (weights, L2-resident)
        acc = __builtin_amdgcn_wmma_f32_16x16x32_f16(false, a, false, b,
                                                     (short)0, acc, false, false);
    }
    // D layout: elem r of lane L -> (M = r + 8*(L>>4), N = L&15)
#pragma unroll
    for (int r = 0; r < 8; ++r)
        tile[(r + 8 * hs) * 132 + wave * 16 + mr] = acc[r];
    __syncthreads();

    float v[8];
#pragma unroll
    for (int q = 0; q < 8; ++q) {
        int e = threadIdx.x * 8 + q;
        int m = e >> 7, c = e & 127;
        float val = tile[m * 132 + c] + bias[col0 + c];
        if (resid) val += resid[(row0 + m) * (size_t)Nc + col0 + c];
        v[q] = val;
    }
    if (g) {  // LayerNorm across the 128 columns (Nc==128 path)
#pragma unroll
        for (int q = 0; q < 8; ++q) {
            int e = threadIdx.x * 8 + q;
            tile[(e >> 7) * 132 + (e & 127)] = v[q];
        }
        __syncthreads();
        if (threadIdx.x < 16) {
            float s = 0.f, ss = 0.f;
            for (int c = 0; c < 128; ++c) {
                float x = tile[threadIdx.x * 132 + c];
                s += x; ss += x * x;
            }
            float mean = s * (1.f / 128.f);
            float var  = ss * (1.f / 128.f) - mean * mean;
            sm[threadIdx.x] = mean;
            sr[threadIdx.x] = rsqrtf(var + 1e-5f);
        }
        __syncthreads();
#pragma unroll
        for (int q = 0; q < 8; ++q) {
            int e = threadIdx.x * 8 + q;
            int m = e >> 7, c = e & 127;
            v[q] = (v[q] - sm[m]) * sr[m] * g[c] + be[c];
        }
    }
#pragma unroll
    for (int q = 0; q < 8; ++q) {
        int e = threadIdx.x * 8 + q;
        int m = e >> 7, c = e & 127;
        float val = v[q];
        if (relu) val = fmaxf(val, 0.f);
        size_t oi = (row0 + m) * (size_t)Nc + col0 + c;
        if (out32) out32[oi] = val;
        if (out16) {
            _Float16 hv = (_Float16)val;
            if (nt) __builtin_nontemporal_store(hv, out16 + oi);  // streaming big tensors
            else    out16[oi] = hv;
        }
    }
}

// ---------------- attention (edge or self), d=16, 64 keys ----------------
// one 64-thread block per (row, head); row in [0,BTN)
// edge_mode=1: K/V indexed per (row, j) [big buffers]; else per (bt*64 + j).
// mask flat index = row*64 + j for both edge_mask and att_mask.
__global__ void k_attn(const _Float16* __restrict__ Qp, const _Float16* __restrict__ Kp,
                       const _Float16* __restrict__ Vp, const int* __restrict__ mask,
                       _Float16* __restrict__ ctx, int edge_mode) {
    __shared__ float sp[64];
    __shared__ float red[64];
    int id  = blockIdx.x;
    int h   = id & 7;
    int row = id >> 3;
    int j   = threadIdx.x;
    size_t kvbase = edge_mode ? ((size_t)row * 64) : (size_t)((row >> 6) << 6);
    const _Float16* q = Qp + (size_t)row * 128 + h * 16;
    const _Float16* k = Kp + (kvbase + j) * 128 + h * 16;
    float s = 0.f;
#pragma unroll
    for (int dd = 0; dd < 16; ++dd) s += (float)q[dd] * (float)k[dd];
    s *= 0.25f;  // 1/sqrt(16)
    if (mask[(size_t)row * 64 + j] == 0) s = -1e9f;
    red[j] = s;
    __syncthreads();
    for (int off = 32; off > 0; off >>= 1) {
        if (j < off) red[j] = fmaxf(red[j], red[j + off]);
        __syncthreads();
    }
    float mx = red[0];
    __syncthreads();
    float p = __expf(s - mx);
    red[j] = p;
    __syncthreads();
    for (int off = 32; off > 0; off >>= 1) {
        if (j < off) red[j] += red[j + off];
        __syncthreads();
    }
    float inv = 1.f / red[0];
    __syncthreads();
    sp[j] = p * inv;
    __syncthreads();
    if (j < 16) {
        int dd = j;
        const _Float16* vb = Vp + kvbase * 128 + h * 16 + dd;
        float acc = 0.f;
        for (int jj = 0; jj < 64; ++jj) acc += sp[jj] * (float)vb[(size_t)jj * 128];
        ctx[(size_t)row * 128 + h * 16 + dd] = (_Float16)acc;
    }
}

// ---------------- final slice x[:, -1] -> [B,N,128] ----------------
__global__ void k_tail(const float* __restrict__ x, float* __restrict__ out) {
    int idx = blockIdx.x * 256 + threadIdx.x;
    if (idx >= BB * NN * 128) return;
    int c = idx & 127;
    int n = (idx >> 7) & 63;
    int b = idx >> 13;
    out[idx] = x[(size_t)(((b * TT + (TT - 1)) * NN) + n) * 128 + c];
}

// ================= host orchestration =================
extern "C" void kernel_launch(void* const* d_in, const int* in_sizes, int n_in,
                              void* d_out, int out_size, void* d_ws, size_t ws_size,
                              hipStream_t stream) {
    (void)in_sizes; (void)n_in; (void)out_size; (void)ws_size;
    const float* agent = (const float*)d_in[0];
    const float* rotf  = (const float*)d_in[1];
    const int*   emask = (const int*)d_in[2];
    const int*   amask = (const int*)d_in[3];

    // att block params in JAX alphabetical pytree order
    struct AttP { const float *be1,*be2,*bf,*bk,*bq,*bv,*fb1,*fb2,*fw1,*fw2,*g1,*g2,*wf,*wk,*wq,*wv; };
    auto getAtt = [&](int b0) {
        AttP p;
        p.be1 = (const float*)d_in[b0+0];  p.be2 = (const float*)d_in[b0+1];
        p.bf  = (const float*)d_in[b0+2];  p.bk  = (const float*)d_in[b0+3];
        p.bq  = (const float*)d_in[b0+4];  p.bv  = (const float*)d_in[b0+5];
        p.fb1 = (const float*)d_in[b0+6];  p.fb2 = (const float*)d_in[b0+7];
        p.fw1 = (const float*)d_in[b0+8];  p.fw2 = (const float*)d_in[b0+9];
        p.g1  = (const float*)d_in[b0+10]; p.g2  = (const float*)d_in[b0+11];
        p.wf  = (const float*)d_in[b0+12]; p.wk  = (const float*)d_in[b0+13];
        p.wq  = (const float*)d_in[b0+14]; p.wv  = (const float*)d_in[b0+15];
        return p;
    };
    AttP edge = getAtt(4);
    const float* fc1_b1 = (const float*)d_in[20];
    const float* fc1_b2 = (const float*)d_in[21];
    const float* fc1_be = (const float*)d_in[22];
    const float* fc1_g  = (const float*)d_in[23];
    const float* fc1_w1 = (const float*)d_in[24];
    const float* fc1_w2 = (const float*)d_in[25];
    const float* fc2_b1 = (const float*)d_in[26];
    const float* fc2_b2 = (const float*)d_in[27];
    const float* fc2_be = (const float*)d_in[28];
    const float* fc2_g  = (const float*)d_in[29];
    const float* fc2_w1 = (const float*)d_in[30];
    const float* fc2_w2 = (const float*)d_in[31];
    AttP sa0 = getAtt(32);
    AttP sa1 = getAtt(48);

    // ---- workspace carve (256B aligned) ----
    char* ws = (char*)d_ws;
    size_t off = 0;
    auto carve = [&](size_t bytes) -> char* {
        char* p = ws + off;
        off += (bytes + 255) & ~(size_t)255;
        return p;
    };
    // f16 transposed weight pool
    _Float16* wt = (_Float16*)carve((size_t)(32768 + 3 * 131072) * 2);
    const size_t FC1W2T = 0, FC2W2T = 16384;
    const size_t WQ = 0, WK = 16384, WV = 32768, WF = 49152, FW1 = 65536, FW2 = 98304;
    auto attWT = [&](int a) { return wt + 32768 + (size_t)a * 131072; };

    float*    xA   = (float*)carve((size_t)BTN * 128 * 4);
    float*    xB   = (float*)carve((size_t)BTN * 128 * 4);
    _Float16* h16  = (_Float16*)carve((size_t)BTN * 128 * 2);
    _Float16* x16  = (_Float16*)carve((size_t)BTN * 128 * 2);
    _Float16* q16  = (_Float16*)carve((size_t)BTN * 128 * 2);
    _Float16* k16s = (_Float16*)carve((size_t)BTN * 128 * 2);
    _Float16* v16s = (_Float16*)carve((size_t)BTN * 128 * 2);
    _Float16* cA16 = (_Float16*)carve((size_t)BTN * 128 * 2);
    _Float16* cB16 = (_Float16*)carve((size_t)BTN * 128 * 2);
    _Float16* ffh  = (_Float16*)carve((size_t)BTN * 256 * 2);
    _Float16* big0 = (_Float16*)carve((size_t)BTNN * 128 * 2);
    _Float16* big1 = (_Float16*)carve((size_t)BTNN * 128 * 2);
    _Float16* big2 = (_Float16*)carve((size_t)BTNN * 128 * 2);

    auto cvt = [&](const float* s, _Float16* d, int K, int Nc) {
        k_cvt<<<dim3((K * Nc + 255) / 256), 256, 0, stream>>>(s, d, K, Nc);
    };
    auto gemm = [&](const _Float16* A, const _Float16* Bt, const float* bias,
                    const float* resid, const float* g, const float* be, int relu, int nt,
                    float* o32, _Float16* o16, int M, int Nc, int K) {
        k_gemm<<<dim3(M / 16, Nc / 128), 256, 0, stream>>>(A, Bt, bias, resid, g, be,
                                                           relu, nt, o32, o16, Nc, K);
    };

    // ---- weight conversion ----
    cvt(fc1_w2, wt + FC1W2T, 128, 128);
    cvt(fc2_w2, wt + FC2W2T, 128, 128);
    const AttP blks[3] = { edge, sa0, sa1 };
    for (int a = 0; a < 3; ++a) {
        _Float16* base = attWT(a);
        cvt(blks[a].wq, base + WQ, 128, 128);
        cvt(blks[a].wk, base + WK, 128, 128);
        cvt(blks[a].wv, base + WV, 128, 128);
        cvt(blks[a].wf, base + WF, 128, 128);
        cvt(blks[a].fw1, base + FW1, 128, 256);
        cvt(blks[a].fw2, base + FW2, 256, 128);
    }

    // ---- x = mlp(agent_features, fc1) ----
    k_fc1_x<<<BTN, 128, 0, stream>>>(agent, fc1_w1, fc1_b1, fc1_g, fc1_be, h16);
    gemm(h16, wt + FC1W2T, fc1_b2, nullptr, nullptr, nullptr, 0, 0, xA, x16, BTN, 128, 128);

    // ---- kv = mlp(edge features, fc2) ----
    k_fc1_kv<<<BTNN, 128, 0, stream>>>(rotf, agent, fc2_w1, fc2_b1, fc2_g, fc2_be, big0);
    gemm(big0, wt + FC2W2T, fc2_b2, nullptr, nullptr, nullptr, 0, 1, nullptr, big1, BTNN, 128, 128);

    // ---- edge attention block ----
    {
        _Float16* wE = attWT(0);
        gemm(x16, wE + WQ, edge.bq, nullptr, nullptr, nullptr, 0, 0, nullptr, q16, BTN, 128, 128);
        gemm(big1, wE + WK, edge.bk, nullptr, nullptr, nullptr, 0, 1, nullptr, big0, BTNN, 128, 128);
        gemm(big1, wE + WV, edge.bv, nullptr, nullptr, nullptr, 0, 1, nullptr, big2, BTNN, 128, 128);
        k_attn<<<BTN * 8, 64, 0, stream>>>(q16, big0, big2, emask, cA16, 1);
        gemm(cA16, wE + WF, edge.bf, xA, edge.g1, edge.be1, 0, 0, xB, cB16, BTN, 128, 128);
        gemm(cB16, wE + FW1, edge.fb1, nullptr, nullptr, nullptr, 1, 0, nullptr, ffh, BTN, 256, 128);
        gemm(ffh, wE + FW2, edge.fb2, xB, edge.g2, edge.be2, 0, 0, xA, x16, BTN, 128, 256);
    }

    // ---- two self-attention blocks ----
    for (int s = 0; s < 2; ++s) {
        const AttP& P = (s == 0) ? sa0 : sa1;
        _Float16* wS = attWT(1 + s);
        gemm(x16, wS + WQ, P.bq, nullptr, nullptr, nullptr, 0, 0, nullptr, q16, BTN, 128, 128);
        gemm(x16, wS + WK, P.bk, nullptr, nullptr, nullptr, 0, 0, nullptr, k16s, BTN, 128, 128);
        gemm(x16, wS + WV, P.bv, nullptr, nullptr, nullptr, 0, 0, nullptr, v16s, BTN, 128, 128);
        k_attn<<<BTN * 8, 64, 0, stream>>>(q16, k16s, v16s, amask, cA16, 0);
        gemm(cA16, wS + WF, P.bf, xA, P.g1, P.be1, 0, 0, xB, cB16, BTN, 128, 128);
        gemm(cB16, wS + FW1, P.fb1, nullptr, nullptr, nullptr, 1, 0, nullptr, ffh, BTN, 256, 128);
        gemm(ffh, wS + FW2, P.fb2, xB, P.g2, P.be2, 0, 0, xA, x16, BTN, 128, 256);
    }

    // ---- output slice ----
    k_tail<<<(BB * NN * 128 + 255) / 256, 256, 0, stream>>>(xA, (float*)d_out);
}
